// HeteroGraphSAGE_627065225618
// MI455X (gfx1250) — matrix operand order repaired and checked by reading, not compiled, not run on hardware
//
#include <hip/hip_runtime.h>
#include <hip/hip_bf16.h>

#define NN 100000
#define DD 128
#define HH 128
#define EE 500000

typedef __attribute__((ext_vector_type(16))) _Float16 v16h;
typedef __attribute__((ext_vector_type(8)))  _Float16 v8h;
typedef __attribute__((ext_vector_type(2)))  _Float16 v2h;
typedef __attribute__((ext_vector_type(8)))  float    v8f;

// ---------------------------------------------------------------------------
// Zero a float region (agg + cnt contiguous), float4 grid-stride.
// ---------------------------------------------------------------------------
__global__ __launch_bounds__(256) void zero_f32(float4* __restrict__ p, int n4) {
    int i = blockIdx.x * 256 + threadIdx.x;
    int stride = gridDim.x * 256;
    float4 z = make_float4(0.f, 0.f, 0.f, 0.f);
    for (; i < n4; i += stride) p[i] = z;
}

// ---------------------------------------------------------------------------
// Weight convert+transpose: W[mat][k][n] f32 -> WT[mat][n][k] f16
// mat = layer*4 + rel, 8 mats of 128x128.
// ---------------------------------------------------------------------------
__global__ __launch_bounds__(256) void cvt_w(const float* __restrict__ W,
                                             _Float16* __restrict__ WT) {
    int idx = blockIdx.x * 256 + threadIdx.x;       // 8*128*128 = 131072 total
    if (idx >= 8 * 128 * 128) return;
    int mat = idx >> 14;
    int rem = idx & 16383;
    int k = rem >> 7;
    int n = rem & 127;
    WT[(mat << 14) + n * 128 + k] = (_Float16)W[idx];
}

// ---------------------------------------------------------------------------
// Edge scatter: one wave per edge, 4 floats per lane, hw float atomics.
// agg[dst] += x[src]; cnt[dst] += 1
// ---------------------------------------------------------------------------
__global__ __launch_bounds__(256) void scatter_edges(const float* __restrict__ x,
                                                     const int* __restrict__ edge,
                                                     float* __restrict__ agg,
                                                     float* __restrict__ cnt) {
    int e = blockIdx.x * 8 + (threadIdx.x >> 5);
    if (e >= EE) return;
    int lane = threadIdx.x & 31;
    int s = edge[e];
    int d = edge[EE + e];
    const float4 v = *(const float4*)(x + s * DD + lane * 4);
    float* ap = agg + d * DD + lane * 4;
    unsafeAtomicAdd(ap + 0, v.x);
    unsafeAtomicAdd(ap + 1, v.y);
    unsafeAtomicAdd(ap + 2, v.z);
    unsafeAtomicAdd(ap + 3, v.w);
    if (lane == 0) unsafeAtomicAdd(cnt + d, 1.0f);
}

// ---------------------------------------------------------------------------
// Fused per-relation SAGE GEMM:
//   accout[i,:] OP= (agg[i,:]/max(cnt,1)) @ Wn + bn + x[i,:] @ Wr
// mode 0: overwrite (first relation)
// mode 1: accumulate
// mode 2: accumulate, then relu(val/4)  (last relation -> layer output)
// Block = 256 threads = 8 waves; block computes a 16x128 output stripe,
// wave w computes the 16x16 tile at columns [16w,16w+16).
// K loop: 4 steps of 32 via V_WMMA_F32_16X16X32_F16 (2 WMMAs/step).
// ---------------------------------------------------------------------------
__global__ __launch_bounds__(256) void sage_rel_gemm(
    const float* __restrict__ xin,      // [NN,128]
    const float* __restrict__ agg,      // [NN,128]
    const float* __restrict__ cnt,      // [NN]
    const _Float16* __restrict__ WnT,   // [128n][128k] f16 (transposed)
    const float* __restrict__ bn,       // [128]
    const _Float16* __restrict__ WrT,   // [128n][128k] f16 (transposed)
    float* __restrict__ accout,         // [NN,128]
    int mode)
{
    __shared__ __align__(16) _Float16 sAx[16][40];  // x tile, f16, padded rows
    __shared__ __align__(16) _Float16 sAg[16][40];  // mean-agg tile
    __shared__ float sInv[16];

    const int tid = threadIdx.x;
    const int wave = tid >> 5;
    const int lane = tid & 31;
    const int rowBase = blockIdx.x * 16;            // 100000/16 = 6250 exact
    const int m = lane & 15;
    const int hiHalf = lane >> 4;                   // 0 | 1
    const int kbA = hiHalf << 4;                    // A k-group: 0 | 16
    const int kbB = hiHalf << 4;                    // B k-group: 0 | 16
    const int n = (wave << 4) + m;                  // output column

    if (tid < 16) {
        float cv = cnt[rowBase + tid];
        sInv[tid] = 1.0f / fmaxf(cv, 1.0f);
    }

    // Staging coordinates: one float2 per thread per tile (512 elems/tile).
    const int idx2 = tid << 1;                      // 0,2,...,510
    const int srr = idx2 >> 5;                      // row 0..15
    const int skk = idx2 & 31;                      // even k within tile

    v8f c = {};

    for (int k0 = 0; k0 < DD; k0 += 32) {
        __syncthreads();   // sInv ready / previous-iteration LDS reads done
        {
            int g = (rowBase + srr) * DD + k0 + skk;
            float2 xv = *(const float2*)(xin + g);
            float2 gv = *(const float2*)(agg + g);
            float inv = sInv[srr];
            v2h xp = { (_Float16)xv.x, (_Float16)xv.y };
            v2h gp = { (_Float16)(gv.x * inv), (_Float16)(gv.y * inv) };
            *(v2h*)&sAx[srr][skk] = xp;
            *(v2h*)&sAg[srr][skk] = gp;
        }
        __syncthreads();

        // A tiles: lane m holds K = kbA..kbA+15 contiguous (matches B)
        //  -> two 16B ds loads each
        v8h xlo = *(const v8h*)&sAx[m][kbA];
        v8h xhi = *(const v8h*)&sAx[m][kbA + 8];
        v8h glo = *(const v8h*)&sAg[m][kbA];
        v8h ghi = *(const v8h*)&sAg[m][kbA + 8];
        v16h ax = __builtin_shufflevector(xlo, xhi, 0,1,2,3,4,5,6,7,8,9,10,11,12,13,14,15);
        v16h ag = __builtin_shufflevector(glo, ghi, 0,1,2,3,4,5,6,7,8,9,10,11,12,13,14,15);

        // B tiles (32x16): lane holds column n, K = k0+kbB .. +15 contiguous
        v16h bN = *(const v16h*)&WnT[n * 128 + k0 + kbB];
        v16h bR = *(const v16h*)&WrT[n * 128 + k0 + kbB];

        c = __builtin_amdgcn_wmma_f32_16x16x32_f16(false, ag, false, bN,
                                                   (short)0, c, false, false);
        c = __builtin_amdgcn_wmma_f32_16x16x32_f16(false, ax, false, bR,
                                                   (short)0, c, false, false);
    }

    // C/D layout: VGPR v -> row = v + 8*hiHalf, col = n
    float bias = bn[n];
    int rtop = rowBase + (hiHalf << 3);
    #pragma unroll
    for (int v = 0; v < 8; ++v) {
        int off = (rtop + v) * HH + n;
        float val = c[v] + bias;
        if (mode >= 1) val += accout[off];
        if (mode == 2) val = fmaxf(val * 0.25f, 0.f);   // HeteroConv mean + ReLU
        accout[off] = val;
    }
}

// ---------------------------------------------------------------------------
// Final classifier: out[i,c] = h[i,:] @ lin_w[:,c] + lin_b[c], c < 16.
// One thread per output element; h row broadcast within half-wave.
// ---------------------------------------------------------------------------
__global__ __launch_bounds__(256) void classify(const float* __restrict__ h,
                                                const float* __restrict__ lw,
                                                const float* __restrict__ lb,
                                                float* __restrict__ out) {
    int gid = blockIdx.x * 256 + threadIdx.x;       // NN*16
    if (gid >= NN * 16) return;
    int i = gid >> 4;
    int cc = gid & 15;
    const float* hr = h + i * HH;
    float s = lb[cc];
    #pragma unroll 8
    for (int k = 0; k < HH; ++k) s = fmaf(hr[k], lw[k * 16 + cc], s);
    out[gid] = s;
}

extern "C" void kernel_launch(void* const* d_in, const int* in_sizes, int n_in,
                              void* d_out, int out_size, void* d_ws, size_t ws_size,
                              hipStream_t stream) {
    const float* x      = (const float*)d_in[0];
    const int*   edges[4] = {(const int*)d_in[1], (const int*)d_in[2],
                             (const int*)d_in[3], (const int*)d_in[4]};
    const float* W_nbr  = (const float*)d_in[5];
    const float* b_nbr  = (const float*)d_in[6];
    const float* W_root = (const float*)d_in[7];
    const float* lin_w  = (const float*)d_in[8];
    const float* lin_b  = (const float*)d_in[9];
    float* out = (float*)d_out;

    // Workspace layout (~155 MB)
    float* agg  = (float*)d_ws;                       // NN*128
    float* cnt  = agg + (size_t)NN * DD;              // NN  (contiguous after agg)
    float* bufA = cnt + NN;                           // NN*128 (layer-0 output)
    float* bufB = bufA + (size_t)NN * DD;             // NN*128 (layer-1 output)
    _Float16* WnT = (_Float16*)(bufB + (size_t)NN * DD);  // 8*128*128 f16
    _Float16* WrT = WnT + 8 * 128 * 128;

    cvt_w<<<512, 256, 0, stream>>>(W_nbr, WnT);
    cvt_w<<<512, 256, 0, stream>>>(W_root, WrT);

    const int zero_n4 = (NN * (DD + 1)) / 4;          // agg + cnt together
    for (int l = 0; l < 2; ++l) {
        const float* xin = (l == 0) ? x : bufA;
        float* acc = (l == 0) ? bufA : bufB;
        for (int r = 0; r < 4; ++r) {
            int mat = l * 4 + r;
            int mode = (r == 0) ? 0 : (r == 3 ? 2 : 1);
            zero_f32<<<2048, 256, 0, stream>>>((float4*)agg, zero_n4);
            scatter_edges<<<EE / 8, 256, 0, stream>>>(xin, edges[r], agg, cnt);
            sage_rel_gemm<<<NN / 16, 256, 0, stream>>>(
                xin, agg, cnt,
                WnT + ((size_t)mat << 14), b_nbr + mat * HH,
                WrT + ((size_t)mat << 14), acc, mode);
        }
    }
    classify<<<(NN * 16) / 256, 256, 0, stream>>>(bufB, lin_w, lin_b, out);
}
